// MultiHeadAttention_51333449122076
// MI455X (gfx1250) — matrix operand order repaired and checked by reading, not compiled, not run on hardware
//
#include <hip/hip_runtime.h>
#include <hip/hip_bf16.h>
#include <cstdint>
#include <cstddef>

#define LSEQ   2048
#define NB     8
#define NH     8
#define DMODEL 512
#define DK     64
#define SROW   2052   // padded f32 row stride for score LDS (row bytes 8208 = 16B-mult)
#define AST    528    // bf16 row stride for staged A (row bytes 1056, 16B aligned chunks)
#define FST    520    // f32 row stride for fc output tile

typedef __attribute__((ext_vector_type(16))) __bf16 v16bf;
typedef __attribute__((ext_vector_type(8)))  __bf16 v8bf;
typedef __attribute__((ext_vector_type(8)))  float  v8f;
typedef __attribute__((ext_vector_type(4)))  float  f32x4;
typedef unsigned int u32;
typedef __attribute__((ext_vector_type(4)))  u32    v4u;

// ---- helpers ----
__device__ __forceinline__ void splitf(float x, __bf16& h, __bf16& l) {
    h = (__bf16)x;                 // native cvt, RNE
    l = (__bf16)(x - (float)h);
}
__device__ __forceinline__ unsigned short bfbits(__bf16 b) {
    union { __bf16 b; unsigned short s; } u; u.b = b; return u.s;
}
__device__ __forceinline__ v16bf cat16(v8bf a, v8bf b) {
    return __builtin_shufflevector(a, b, 0,1,2,3,4,5,6,7,8,9,10,11,12,13,14,15);
}
// fp32-emulating product: (ah+al)*(bh+bl) ~= ah*bh + ah*bl + al*bh
__device__ __forceinline__ v8f wmma3(v16bf ah, v16bf al, v16bf bh, v16bf bl, v8f c) {
    c = __builtin_amdgcn_wmma_f32_16x16x32_bf16(false, ah, false, bh, (short)0, c, false, false);
    c = __builtin_amdgcn_wmma_f32_16x16x32_bf16(false, ah, false, bl, (short)0, c, false, false);
    c = __builtin_amdgcn_wmma_f32_16x16x32_bf16(false, al, false, bh, (short)0, c, false, false);
    return c;
}
// unpack 16 packed (hi|lo<<16) dwords into hi/lo v16bf via v_perm_b32
union bfpack { v16bf v; u32 d[8]; };
__device__ __forceinline__ void unpack16(v4u a0, v4u a1, v4u a2, v4u a3,
                                         v16bf& hi, v16bf& lo) {
#if __has_builtin(__builtin_amdgcn_perm)
    bfpack H, L;
    H.d[0] = __builtin_amdgcn_perm(a0[1], a0[0], 0x05040100u);
    H.d[1] = __builtin_amdgcn_perm(a0[3], a0[2], 0x05040100u);
    H.d[2] = __builtin_amdgcn_perm(a1[1], a1[0], 0x05040100u);
    H.d[3] = __builtin_amdgcn_perm(a1[3], a1[2], 0x05040100u);
    H.d[4] = __builtin_amdgcn_perm(a2[1], a2[0], 0x05040100u);
    H.d[5] = __builtin_amdgcn_perm(a2[3], a2[2], 0x05040100u);
    H.d[6] = __builtin_amdgcn_perm(a3[1], a3[0], 0x05040100u);
    H.d[7] = __builtin_amdgcn_perm(a3[3], a3[2], 0x05040100u);
    L.d[0] = __builtin_amdgcn_perm(a0[1], a0[0], 0x07060302u);
    L.d[1] = __builtin_amdgcn_perm(a0[3], a0[2], 0x07060302u);
    L.d[2] = __builtin_amdgcn_perm(a1[1], a1[0], 0x07060302u);
    L.d[3] = __builtin_amdgcn_perm(a1[3], a1[2], 0x07060302u);
    L.d[4] = __builtin_amdgcn_perm(a2[1], a2[0], 0x07060302u);
    L.d[5] = __builtin_amdgcn_perm(a2[3], a2[2], 0x07060302u);
    L.d[6] = __builtin_amdgcn_perm(a3[1], a3[0], 0x07060302u);
    L.d[7] = __builtin_amdgcn_perm(a3[3], a3[2], 0x07060302u);
    hi = H.v; lo = L.v;
#else
    bfpack H, L;
    const v4u* src[4] = { &a0, &a1, &a2, &a3 };
#pragma unroll
    for (int g = 0; g < 4; ++g) {
        u32 x0 = (*src[g])[0], x1 = (*src[g])[1], x2 = (*src[g])[2], x3 = (*src[g])[3];
        H.d[2*g]   = (x0 & 0xffffu) | (x1 << 16);
        H.d[2*g+1] = (x2 & 0xffffu) | (x3 << 16);
        L.d[2*g]   = (x0 >> 16) | (x1 & 0xffff0000u);
        L.d[2*g+1] = (x2 >> 16) | (x3 & 0xffff0000u);
    }
    hi = H.v; lo = L.v;
#endif
}

// ===================== Prep: one-time bf16 hi/lo conversion =====================
__global__ __launch_bounds__(256)
void prep_kernel(const float* __restrict__ k, const float* __restrict__ v,
                 const float* __restrict__ fc_w,
                 __bf16* __restrict__ khi, __bf16* __restrict__ klo,
                 __bf16* __restrict__ vthi, __bf16* __restrict__ vtlo,
                 __bf16* __restrict__ whi, __bf16* __restrict__ wlo) {
    const size_t i = (size_t)blockIdx.x * 256 + threadIdx.x;
    const size_t NKV = (size_t)LSEQ * NB * DMODEL;
    if (i < NKV) {
        splitf(k[i], khi[i], klo[i]);                      // coalesced r/w
        // transposed V: i = (n*64+col)*2048 + key  (coalesced writes along key)
        const size_t key = i & 2047;
        const size_t rest = i >> 11;                       // n*64+col
        const size_t col = rest & 63, n = rest >> 6;
        const size_t b = n >> 3, h = n & 7;
        splitf(v[(key * NB + b) * DMODEL + h * DK + col], vthi[i], vtlo[i]);
    }
    if (i < (size_t)DMODEL * DMODEL) splitf(fc_w[i], whi[i], wlo[i]);
}

// ======================= Kernel A: attention ============================
// grid = (L/16, B*H), block = 128 (4 wave32).  LDS ~135 KB -> 2 WGs/WGP.
__global__ __launch_bounds__(128)
void attn_kernel(const float* __restrict__ q, const unsigned char* __restrict__ mask,
                 const __bf16* __restrict__ khi, const __bf16* __restrict__ klo,
                 const __bf16* __restrict__ vthi, const __bf16* __restrict__ vtlo,
                 float* __restrict__ attn_out, float* __restrict__ merged) {
    extern __shared__ float smem[];
    float* slds = smem;                // [16][SROW]
    float* qlds = smem + 16 * SROW;    // [16][DK]

    const int qb   = blockIdx.x;
    const int n    = blockIdx.y;       // b-major head-batch: n = b*H + h
    const int b    = n >> 3;
    const int h    = n & 7;
    const int maskb = n & 7;           // faithful to mask.tile(H,1,1): index = n % B
    const int tid  = threadIdx.x;
    const int lane = tid & 31;
    const int wv   = tid >> 5;
    const int mrow = lane & 15;
    const int half = lane >> 4;
    const int nc   = lane & 15;

    // stage Q tile [16][64]
    for (int i = tid; i < 16 * DK; i += 128) {
        int mm = i >> 6, d = i & 63;
        qlds[i] = q[((size_t)(qb * 16 + mm) * NB + b) * DMODEL + h * DK + d];
    }
    __syncthreads();

    // per-lane A operand (Q rows), two 32-wide dk chunks, hi/lo split (one-time)
    v16bf aqh[2], aql[2];
#pragma unroll
    for (int c = 0; c < 2; ++c) {
        const float* base = &qlds[mrow * DK + c * 32 + half * 16];
#pragma unroll
        for (int i = 0; i < 8; ++i) {
            __bf16 hh, ll;
            splitf(base[i], hh, ll);       aqh[c][i] = hh;     aql[c][i] = ll;
            splitf(base[8 + i], hh, ll);  aqh[c][8 + i] = hh; aql[c][8 + i] = ll;
        }
    }

    // ---- S = (Q K^T)/8, masked, into LDS.  Wave wv owns key cols [wv*512, +512) ----
    const size_t kstep = (size_t)16 * NB * DMODEL;   // 16 keys ahead
    for (int kt = 0; kt < 32; ++kt) {
        const int k0 = wv * 512 + kt * 16;
        const size_t kbase = ((size_t)(k0 + nc) * NB + b) * DMODEL + h * DK;
        if (kt < 31) {                                   // gfx1250 global_prefetch_b8
            __builtin_prefetch(khi + kbase + kstep, 0, 3);
            __builtin_prefetch(klo + kbase + kstep, 0, 3);
        }
        v8f acc = {};
#pragma unroll
        for (int c = 0; c < 2; ++c) {
            const __bf16* ph = khi + kbase + c * 32 + half * 16;
            const __bf16* pl = klo + kbase + c * 32 + half * 16;
            v16bf bh_ = cat16(*(const v8bf*)ph, *(const v8bf*)(ph + 8));
            v16bf bl_ = cat16(*(const v8bf*)pl, *(const v8bf*)(pl + 8));
            acc = wmma3(aqh[c], aql[c], bh_, bl_, acc);
        }
#pragma unroll
        for (int r = 0; r < 8; ++r) {
            const int row = r + half * 8;        // C/D layout: M = r + 8*(lane>>4)
            const int lq  = qb * 16 + row;
            float s = acc[r] * 0.125f;           // 1/sqrt(dk)
            if (mask[((size_t)maskb * LSEQ + lq) * LSEQ + k0 + nc]) s = -1e9f;
            slds[row * SROW + k0 + nc] = s;
        }
    }
    __syncthreads();

    // ---- row softmax, fully float4-vectorized; write attn once (b128 stores);
    //      repack P as (hi|lo<<16) in place (same dwords just read: race-free) ----
    for (int rr = wv * 4; rr < wv * 4 + 4; ++rr) {
        float* srow = slds + rr * SROW;
        u32*   srowu = (u32*)srow;
        float mx = -3.4e38f;
#pragma unroll 4
        for (int i = 0; i < 16; ++i) {
            const f32x4 xv = *(const f32x4*)&srow[lane * 4 + i * 128];
            mx = fmaxf(mx, fmaxf(fmaxf(xv[0], xv[1]), fmaxf(xv[2], xv[3])));
        }
#pragma unroll
        for (int off = 16; off > 0; off >>= 1) mx = fmaxf(mx, __shfl_xor(mx, off, 32));
        float sum = 0.f;
#pragma unroll 4
        for (int i = 0; i < 16; ++i) {
            const int j = lane * 4 + i * 128;
            f32x4 xv = *(const f32x4*)&srow[j];
            f32x4 ev;
            ev[0] = __expf(xv[0] - mx); ev[1] = __expf(xv[1] - mx);
            ev[2] = __expf(xv[2] - mx); ev[3] = __expf(xv[3] - mx);
            *(f32x4*)&srow[j] = ev;
            sum += (ev[0] + ev[1]) + (ev[2] + ev[3]);
        }
#pragma unroll
        for (int off = 16; off > 0; off >>= 1) sum += __shfl_xor(sum, off, 32);
        const float inv = 1.0f / sum;
        const size_t abase = ((size_t)n * LSEQ + qb * 16 + rr) * LSEQ;
#pragma unroll 4
        for (int i = 0; i < 16; ++i) {
            const int j = lane * 4 + i * 128;
            f32x4 pv = *(const f32x4*)&srow[j];
            pv = pv * inv;
            *(f32x4*)&attn_out[abase + j] = pv;            // 512B per wave-instr
            v4u pk;
#pragma unroll
            for (int e = 0; e < 4; ++e) {
                __bf16 ph, pl; splitf(pv[e], ph, pl);      // convert ONCE per element
                pk[e] = (u32)bfbits(ph) | ((u32)bfbits(pl) << 16);
            }
            *(v4u*)&srowu[j] = pk;
        }
    }
    __syncthreads();

    // ---- O = P @ V.  Wave wv owns dk output cols [wv*16, +16). ----
    v8f oacc = {};
    const u32* su = (const u32*)slds;
    const size_t vtb = ((size_t)n * DK + wv * 16 + nc) * LSEQ;   // [n][col][*]
    for (int kc = 0; kc < 64; ++kc) {
        const int kb0 = kc * 32;
        // A operand: packed P, 4 aligned uint4 LDS loads + v_perm unpack
        const u32* pa = &su[mrow * SROW + kb0 + half * 16];
        v16bf pah, pal;
        unpack16(*(const v4u*)pa, *(const v4u*)(pa + 4),
                 *(const v4u*)(pa + 8), *(const v4u*)(pa + 12), pah, pal);
        // B operand: pre-transposed bf16 V, contiguous along key
        const __bf16* bh = vthi + vtb + kb0 + half * 16;
        const __bf16* bl = vtlo + vtb + kb0 + half * 16;
        v16bf vbh = cat16(*(const v8bf*)bh, *(const v8bf*)(bh + 8));
        v16bf vbl = cat16(*(const v8bf*)bl, *(const v8bf*)(bl + 8));
        oacc = wmma3(pah, pal, vbh, vbl, oacc);
    }
    // faithful head-merge bug: flat n reinterpreted as (h'=n/B, b'=n%B)
#pragma unroll
    for (int r = 0; r < 8; ++r) {
        const int lq = qb * 16 + r + half * 8;
        merged[((size_t)h * LSEQ + lq) * DMODEL + b * DK + wv * 16 + nc] = oacc[r];
    }
}

// ================= Kernel B: FC + residual + LayerNorm ==================
__global__ __launch_bounds__(128)
void fc_ln_kernel(const float* __restrict__ q, const float* __restrict__ merged,
                  const __bf16* __restrict__ whi, const __bf16* __restrict__ wlo,
                  const float* __restrict__ fc_b,
                  const float* __restrict__ ln_g, const float* __restrict__ ln_b,
                  float* __restrict__ y) {
    extern __shared__ unsigned char smemraw[];
    __bf16* ahi = (__bf16*)smemraw;            // [16][AST]
    __bf16* alo = ahi + 16 * AST;              // [16][AST]
    float*  fout = (float*)(alo + 16 * AST);   // [16][FST]

    const int j0  = blockIdx.x * 16;
    const int bb  = j0 >> 11;
    const int lq0 = j0 & 2047;
    const int tid = threadIdx.x, lane = tid & 31, wv = tid >> 5;
    const int mrow = lane & 15, half = lane >> 4, nc = lane & 15;

    // stage A rows (hi/lo bf16) once; reused by every output col tile
    for (int i = tid; i < 16 * DMODEL; i += 128) {
        int mm = i >> 9, d = i & 511;
        float x = merged[((size_t)bb * LSEQ + lq0 + mm) * DMODEL + d];
        __bf16 hh, ll; splitf(x, hh, ll);
        ahi[mm * AST + d] = hh;
        alo[mm * AST + d] = ll;
    }
    __syncthreads();

    for (int t = 0; t < 8; ++t) {
        const int ct   = t * 4 + wv;
        const int colj = ct * 16 + nc;
        const __bf16* wh = whi + (size_t)colj * DMODEL;   // out = A @ fc_w^T
        const __bf16* wl = wlo + (size_t)colj * DMODEL;
        v8f acc = {};
        for (int cc = 0; cc < 16; ++cc) {
            const int cb = cc * 32;
            v16bf mah = cat16(*(const v8bf*)&ahi[mrow * AST + cb + half * 16],
                              *(const v8bf*)&ahi[mrow * AST + cb + half * 16 + 8]);
            v16bf mal = cat16(*(const v8bf*)&alo[mrow * AST + cb + half * 16],
                              *(const v8bf*)&alo[mrow * AST + cb + half * 16 + 8]);
            v16bf wbh = cat16(*(const v8bf*)(wh + cb + half * 16),
                              *(const v8bf*)(wh + cb + half * 16 + 8));
            v16bf wbl = cat16(*(const v8bf*)(wl + cb + half * 16),
                              *(const v8bf*)(wl + cb + half * 16 + 8));
            acc = wmma3(mah, mal, wbh, wbl, acc);
        }
        const float bias = fc_b[colj];
#pragma unroll
        for (int r = 0; r < 8; ++r)
            fout[(r + half * 8) * FST + ct * 16 + nc] = acc[r] + bias;
    }
    __syncthreads();

    // residual add ([B,L,D] -> [L,B,D] transpose) + LayerNorm, fused in LDS
    for (int rr = wv * 4; rr < wv * 4 + 4; ++rr) {
        const int lq = lq0 + rr;
        const float* res  = q + ((size_t)lq * NB + bb) * DMODEL;
        const float* frow = fout + rr * FST;
        float xs[16];
        float s1 = 0.f, s2 = 0.f;
#pragma unroll
        for (int ii = 0; ii < 16; ++ii) {
            const int d = lane + ii * 32;
            float x = frow[d] + res[d];
            xs[ii] = x; s1 += x; s2 += x * x;
        }
#pragma unroll
        for (int off = 16; off > 0; off >>= 1) {
            s1 += __shfl_xor(s1, off, 32);
            s2 += __shfl_xor(s2, off, 32);
        }
        const float mean = s1 * (1.0f / DMODEL);
        const float var  = s2 * (1.0f / DMODEL) - mean * mean;
        const float rstd = rsqrtf(var + 1e-5f);
        float* yrow = y + ((size_t)lq * NB + bb) * DMODEL;
#pragma unroll
        for (int ii = 0; ii < 16; ++ii) {
            const int d = lane + ii * 32;
            yrow[d] = (xs[ii] - mean) * rstd * ln_g[d] + ln_b[d];
        }
    }
}

// ============================ launch ====================================
extern "C" void kernel_launch(void* const* d_in, const int* in_sizes, int n_in,
                              void* d_out, int out_size, void* d_ws, size_t ws_size,
                              hipStream_t stream) {
    const float*         q    = (const float*)d_in[0];
    const float*         k    = (const float*)d_in[1];
    const float*         v    = (const float*)d_in[2];
    const unsigned char* mask = (const unsigned char*)d_in[3];
    const float*         fc_w = (const float*)d_in[4];
    const float*         fc_b = (const float*)d_in[5];
    const float*         ln_g = (const float*)d_in[6];
    const float*         ln_b = (const float*)d_in[7];

    float* y    = (float*)d_out;                                  // [L,B,D]
    float* attn = y + (size_t)LSEQ * NB * DMODEL;                 // [B*H,L,L]

    // workspace layout (bytes, all 16B aligned)
    const size_t NKV = (size_t)LSEQ * NB * DMODEL;                // 8,388,608 elems
    unsigned char* ws = (unsigned char*)d_ws;
    float*  merged = (float*)ws;                                  // 33,554,432 B
    __bf16* khi  = (__bf16*)(ws + 33554432);                      // 16,777,216 B
    __bf16* klo  = (__bf16*)(ws + 50331648);
    __bf16* vthi = (__bf16*)(ws + 67108864);
    __bf16* vtlo = (__bf16*)(ws + 83886080);
    __bf16* whi  = (__bf16*)(ws + 100663296);                     // 524,288 B
    __bf16* wlo  = (__bf16*)(ws + 101187584);                     // total ~97 MB

    prep_kernel<<<dim3((unsigned)(NKV / 256)), dim3(256), 0, stream>>>(
        k, v, fc_w, khi, klo, vthi, vtlo, whi, wlo);

    // qblock fastest in launch order -> per-head K/V stay hot in L2
    dim3 gA(LSEQ / 16, NB * NH);
    size_t ldsA = (size_t)(16 * SROW + 16 * DK) * sizeof(float);  // ~135 KB
    attn_kernel<<<gA, dim3(128), ldsA, stream>>>(q, mask, khi, klo, vthi, vtlo, attn, merged);

    dim3 gB((LSEQ * NB) / 16);
    size_t ldsB = (size_t)2 * 16 * AST * sizeof(__bf16) + (size_t)16 * FST * sizeof(float);
    fc_ln_kernel<<<gB, dim3(128), ldsB, stream>>>(q, merged, whi, wlo, fc_b, ln_g, ln_b, y);
}
